// DecoderBlock_66408784330765
// MI455X (gfx1250) — matrix operand order repaired and checked by reading, not compiled, not run on hardware
//
#include <hip/hip_runtime.h>
#include <hip/hip_bf16.h>

// ---------------------------------------------------------------------------
// DecoderBlock for MI455X (gfx1250): bf16 WMMA GEMMs + TDM-staged tiles +
// flash attention.  B=2, S=2048, D=1024, H=16, DH=64, E=8, FH=4096, T=4096.
// ---------------------------------------------------------------------------

typedef __attribute__((ext_vector_type(16))) __bf16 v16bf;
typedef __attribute__((ext_vector_type(8)))  float  v8f;
typedef __attribute__((ext_vector_type(4)))  unsigned int u32x4;
typedef __attribute__((ext_vector_type(8)))  int i32x8;
typedef __attribute__((ext_vector_type(4)))  int i32x4;

union AF { v16bf v; unsigned int u[8]; };

__device__ __forceinline__ unsigned short f2bf(float f) {
  unsigned int u = __float_as_uint(f);
  u += 0x7FFFu + ((u >> 16) & 1u);           // round-to-nearest-even
  return (unsigned short)(u >> 16);
}
__device__ __forceinline__ float bf2f(unsigned short h) {
  return __uint_as_float(((unsigned int)h) << 16);
}
__device__ __forceinline__ unsigned int pk2(float lo, float hi) {
  return (unsigned int)f2bf(lo) | ((unsigned int)f2bf(hi) << 16);
}
__device__ __forceinline__ v8f wmma_bf16(const AF& a, const AF& b, v8f c) {
  return __builtin_amdgcn_wmma_f32_16x16x32_bf16(
      false, a.v, false, b.v, (short)0, c, false, false);
}
__device__ __forceinline__ float gelu_tanh(float v) {
  float c = v + 0.044715f * v * v * v;
  return 0.5f * v * (1.0f + tanhf(0.7978845608028654f * c));
}

// ---------------------------------------------------------------------------
// TDM: 2D tile (tile_d0 x tile_d1, bf16) from global -> LDS, packed rows.
// D# layout per CDNA5 ISA. Groups 2/3 zero (2D tensor).
// This toolchain's 6-arg builtin:
//   (u32x4 g0, i32x8 g1, i32x4 g2, i32x4 g3, i32x8 g4, i32 cpol)
// ---------------------------------------------------------------------------
__device__ __forceinline__ void tdm_load_bf16_2d(unsigned lds_off,
                                                 const void* gaddr,
                                                 unsigned tensor_d0,
                                                 unsigned tensor_d1,
                                                 unsigned tile_d0,
                                                 unsigned tile_d1,
                                                 unsigned stride_elems) {
  unsigned long long ga = (unsigned long long)gaddr;
  u32x4 g0;
  g0.x = 1u;                                       // count=1, user descriptor
  g0.y = lds_off;                                  // lds_addr (bytes)
  g0.z = (unsigned)(ga & 0xffffffffu);             // global_addr[31:0]
  g0.w = (unsigned)((ga >> 32) & 0x01ffffffu)      // global_addr[56:32]
         | (2u << 30);                             // type = 2 ("image")
  i32x8 g1;
  g1[0] = (int)(1u << 16);                         // data_size=1 -> 2 bytes
  g1[1] = (int)((tensor_d0 & 0xffffu) << 16);      // tensor_dim0[15:0]
  g1[2] = (int)((tensor_d0 >> 16) | ((tensor_d1 & 0xffffu) << 16));
  g1[3] = (int)((tensor_d1 >> 16) | (tile_d0 << 16));   // tile_dim0
  g1[4] = (int)(tile_d1 & 0xffffu);                // tile_dim1 (tile_dim2=0)
  g1[5] = (int)stride_elems;                       // tensor_dim0_stride[31:0]
  g1[6] = 0;                                       // stride hi, dim1_stride lo
  g1[7] = 0;
  i32x4 z4 = {0, 0, 0, 0};
  i32x8 z8 = {0, 0, 0, 0, 0, 0, 0, 0};
  __builtin_amdgcn_tensor_load_to_lds(g0, g1, z4, z4, z8, 0);
}

// ---------------------------------------------------------------------------
// LayerNorm: one block per row (D=1024, 256 threads, 4 elems/thread) -> bf16
// ---------------------------------------------------------------------------
__global__ __launch_bounds__(256)
void k_ln(const float* __restrict__ x, const float* __restrict__ gam,
          const float* __restrict__ bet, unsigned short* __restrict__ o) {
  const int row = blockIdx.x, tid = threadIdx.x;
  const float* xr = x + (size_t)row * 1024;
  float v[4]; float s = 0.f;
#pragma unroll
  for (int i = 0; i < 4; ++i) { v[i] = xr[tid + i * 256]; s += v[i]; }
  __shared__ float red[8];
#pragma unroll
  for (int m = 16; m >= 1; m >>= 1) s += __shfl_xor(s, m, 32);
  if ((tid & 31) == 0) red[tid >> 5] = s;
  __syncthreads();
  float mu = 0.f;
#pragma unroll
  for (int i = 0; i < 8; ++i) mu += red[i];
  mu *= (1.0f / 1024.0f);
  float q = 0.f;
#pragma unroll
  for (int i = 0; i < 4; ++i) { float d = v[i] - mu; q += d * d; }
  __syncthreads();
#pragma unroll
  for (int m = 16; m >= 1; m >>= 1) q += __shfl_xor(q, m, 32);
  if ((tid & 31) == 0) red[tid >> 5] = q;
  __syncthreads();
  float var = 0.f;
#pragma unroll
  for (int i = 0; i < 8; ++i) var += red[i];
  var *= (1.0f / 1024.0f);
  float rstd = rsqrtf(var + 1e-6f);
#pragma unroll
  for (int i = 0; i < 4; ++i) {
    int c = tid + i * 256;
    o[(size_t)row * 1024 + c] = f2bf((v[i] - mu) * rstd * gam[c] + bet[c]);
  }
}

// ---------------------------------------------------------------------------
// Generic bf16 WMMA GEMM:  out = epi(A[M,K](bf16) @ Bw[K,N](f32->bf16) + bias)
// Block tile 64x128, BK=32, 256 threads = 8 waves (2x4), wave tile 32x32.
// A tiles DMA'd by the Tensor Data Mover into double-buffered LDS (wave 0
// issues, s_wait_tensorcnt + workgroup barrier synchronize); B tiles converted
// f32->bf16 and K-pair packed by VALU.
// EPI: 0=bf16, 1=bf16*scale, 2=f32+resid, 3=gelu->bf16, 4=moe accumulate
// ---------------------------------------------------------------------------
#define GBM 64
#define GBN 128
#define GBK 32

template <int EPI>
__global__ __launch_bounds__(256)
void k_gemm(const unsigned short* __restrict__ A, int lda,
            const float* __restrict__ Bw, int ldb,
            const float* __restrict__ bias, void* __restrict__ outp,
            const float* __restrict__ resid,
            const float* __restrict__ combine, int cidx,
            float scale, int N, int Kd) {
  __shared__ __align__(16) unsigned int sA[2][GBM * GBK / 2];  // 2 x 64x16 u32
  __shared__ __align__(16) unsigned int sB[(GBK / 2) * GBN];   // 16 kpairs x128
  const int tid = threadIdx.x;
  const int lane = tid & 31, wave = tid >> 5;
  const int half = lane >> 4, r = lane & 15;
  const int wm = wave >> 2, wn = wave & 3;
  const int row0 = blockIdx.y * GBM;
  const int col0 = blockIdx.x * GBN;

  v8f acc[2][2];
#pragma unroll
  for (int i = 0; i < 2; ++i)
#pragma unroll
    for (int j = 0; j < 2; ++j) acc[i][j] = (v8f){0, 0, 0, 0, 0, 0, 0, 0};

  const int pr = tid >> 4, cg = tid & 15;  // B: 16 kpairs x 16 colgroups of 8

  const unsigned short* Arow = A + (size_t)row0 * lda;
  const unsigned ldsA[2] = {(unsigned)(uintptr_t)&sA[0][0],
                            (unsigned)(uintptr_t)&sA[1][0]};
  const int nk = Kd / GBK;

  // Prime the pipeline: DMA first A tile (64 rows x 32 bf16, row stride lda).
  if (wave == 0)
    tdm_load_bf16_2d(ldsA[0], Arow, (unsigned)Kd, (unsigned)(GBM),
                     GBK, GBM, (unsigned)lda);

  for (int ik = 0; ik < nk; ++ik) {
    const int k0 = ik * GBK;
    const int cur = ik & 1;
    // ---- stage B tile, convert f32 -> bf16, pack K-pairs ----
    const float* bg0 = Bw + (size_t)(k0 + 2 * pr) * ldb + col0 + cg * 8;
    const float* bg1 = bg0 + ldb;
    if (ik + 1 < nk) __builtin_prefetch(bg0 + (size_t)GBK * ldb, 0, 3);
    float4 x0 = ((const float4*)bg0)[0], x1 = ((const float4*)bg0)[1];
    float4 y0 = ((const float4*)bg1)[0], y1 = ((const float4*)bg1)[1];
    uint4 p0, p1;
    p0.x = pk2(x0.x, y0.x); p0.y = pk2(x0.y, y0.y);
    p0.z = pk2(x0.z, y0.z); p0.w = pk2(x0.w, y0.w);
    p1.x = pk2(x1.x, y1.x); p1.y = pk2(x1.y, y1.y);
    p1.z = pk2(x1.z, y1.z); p1.w = pk2(x1.w, y1.w);
    *(uint4*)&sB[pr * GBN + cg * 8] = p0;
    *(uint4*)&sB[pr * GBN + cg * 8 + 4] = p1;
    // ---- prefetch next A tile via TDM into the other buffer ----
    if (wave == 0) {
      if (ik + 1 < nk) {
        tdm_load_bf16_2d(ldsA[cur ^ 1], Arow + k0 + GBK, (unsigned)Kd,
                         (unsigned)(GBM), GBK, GBM, (unsigned)lda);
        // current tile must be resident; prefetch may stay in flight
        __builtin_amdgcn_s_wait_tensorcnt(1);
      } else {
        __builtin_amdgcn_s_wait_tensorcnt(0);
      }
    }
    __syncthreads();

    const unsigned int* sAc = &sA[cur][0];
    AF a[2], b[2];
#pragma unroll
    for (int mi = 0; mi < 2; ++mi) {
      int m = wm * 32 + mi * 16 + r;
      int base = m * 16 + half * 4;
#pragma unroll
      for (int j = 0; j < 4; ++j) {
        a[mi].u[j] = sAc[base + j];
        a[mi].u[4 + j] = sAc[base + 8 + j];
      }
    }
#pragma unroll
    for (int ni = 0; ni < 2; ++ni) {
      int n = wn * 32 + ni * 16 + r;
#pragma unroll
      for (int v = 0; v < 8; ++v) b[ni].u[v] = sB[(half * 4 + (v & 3) + (v >> 2) * 8) * GBN + n];
    }
#pragma unroll
    for (int mi = 0; mi < 2; ++mi)
#pragma unroll
      for (int ni = 0; ni < 2; ++ni)
        acc[mi][ni] = wmma_bf16(a[mi], b[ni], acc[mi][ni]);
    __syncthreads();
  }

  // ---- epilogue ----
  const int gmB = row0 + wm * 32, gnB = col0 + wn * 32;
#pragma unroll
  for (int mi = 0; mi < 2; ++mi)
#pragma unroll
    for (int ni = 0; ni < 2; ++ni) {
#pragma unroll
      for (int e = 0; e < 8; ++e) {
        int gm = gmB + mi * 16 + half * 8 + e;
        int gn = gnB + ni * 16 + r;
        float val = acc[mi][ni][e] + bias[gn];
        size_t idx = (size_t)gm * N + gn;
        if constexpr (EPI == 0) {
          ((unsigned short*)outp)[idx] = f2bf(val);
        } else if constexpr (EPI == 1) {
          ((unsigned short*)outp)[idx] = f2bf(val * scale);
        } else if constexpr (EPI == 2) {
          ((float*)outp)[idx] = val + resid[idx];
        } else if constexpr (EPI == 3) {
          ((unsigned short*)outp)[idx] = f2bf(gelu_tanh(val));
        } else {
          float c = combine[(size_t)gm * 8 + cidx];
          ((float*)outp)[idx] += c * val;
        }
      }
    }
}

// ---------------------------------------------------------------------------
// Flash attention: block = 64 q rows for one (b,h); 4 waves x 16 rows.
// kv processed in blocks of 32. q/k/v are bf16 [T, H*DH]; out bf16 same.
// q is pre-scaled by 1/sqrt(DH) in its GEMM epilogue.
// ---------------------------------------------------------------------------
__global__ __launch_bounds__(128)
void k_attn(const unsigned short* __restrict__ qb,
            const unsigned short* __restrict__ kb,
            const unsigned short* __restrict__ vb,
            unsigned short* __restrict__ ob) {
  __shared__ __align__(16) unsigned int uK[32 * 32];      // [kv][dpair]
  __shared__ __align__(16) unsigned int uV[16 * 64];      // [kvpair][d]
  __shared__ __align__(16) unsigned int uP[4 * 16 * 16];  // per-wave P 16x32
  const int S = 2048;
  const int tid = threadIdx.x, lane = tid & 31, wave = tid >> 5;
  const int half = lane >> 4, r = lane & 15;
  const int qbase = blockIdx.x * 64, h = blockIdx.y, b = blockIdx.z;

  // Q A-fragments (2 chunks of K=32 over DH=64), held in registers.
  AF qf[2];
  {
    const int qrow = qbase + wave * 16 + r;
    const unsigned int* qg =
        (const unsigned int*)qb + (size_t)(b * S + qrow) * 512 + h * 32;
#pragma unroll
    for (int c = 0; c < 2; ++c)
#pragma unroll
      for (int j = 0; j < 4; ++j) {
        qf[c].u[j] = qg[c * 16 + half * 4 + j];
        qf[c].u[4 + j] = qg[c * 16 + 8 + half * 4 + j];
      }
  }

  v8f acc[4];
#pragma unroll
  for (int i = 0; i < 4; ++i) acc[i] = (v8f){0, 0, 0, 0, 0, 0, 0, 0};
  float mrow[8], lrow[8];
#pragma unroll
  for (int e = 0; e < 8; ++e) { mrow[e] = -3.0e38f; lrow[e] = 0.f; }

  const int jmax = qbase + 64;
  for (int jb = 0; jb < jmax; jb += 32) {
    // ---- stage K tile: 32 kv rows x 64 d (d-pairs contiguous) ----
    {
      int row = tid >> 2, part = tid & 3;
      const unsigned int* kg = (const unsigned int*)kb +
                               (size_t)(b * S + jb + row) * 512 + h * 32 + part * 8;
      uint4 t0 = ((const uint4*)kg)[0];
      uint4 t1 = ((const uint4*)kg)[1];
      *(uint4*)&uK[row * 32 + part * 8] = t0;
      *(uint4*)&uK[row * 32 + part * 8 + 4] = t1;
    }
    // ---- stage V tile: kv-pair packed [16][64] ----
    {
      int prv = tid >> 3, dg = tid & 7;
      const unsigned int* vg0 = (const unsigned int*)vb +
                                (size_t)(b * S + jb + 2 * prv) * 512 + h * 32 + dg * 4;
      const unsigned int* vg1 = vg0 + 512;
      uint4 lo = *(const uint4*)vg0, hi = *(const uint4*)vg1;
      unsigned int lu[4] = {lo.x, lo.y, lo.z, lo.w};
      unsigned int hu[4] = {hi.x, hi.y, hi.z, hi.w};
      unsigned int o[8];
#pragma unroll
      for (int j = 0; j < 8; ++j) {
        unsigned int a = (lu[j >> 1] >> (16 * (j & 1))) & 0xffffu;
        unsigned int c = (hu[j >> 1] >> (16 * (j & 1))) & 0xffffu;
        o[j] = a | (c << 16);
      }
      uint4 o0 = {o[0], o[1], o[2], o[3]}, o1 = {o[4], o[5], o[6], o[7]};
      *(uint4*)&uV[prv * 64 + dg * 8] = o0;
      *(uint4*)&uV[prv * 64 + dg * 8 + 4] = o1;
    }
    __syncthreads();

    // ---- scores S = Q @ K^T : two 16x16 C-tiles per wave ----
    v8f sc[2];
#pragma unroll
    for (int nt = 0; nt < 2; ++nt) {
      sc[nt] = (v8f){0, 0, 0, 0, 0, 0, 0, 0};
      AF bf_;
#pragma unroll
      for (int v = 0; v < 8; ++v)
        bf_.u[v] = uK[(nt * 16 + r) * 32 + half * 4 + (v & 3) + (v >> 2) * 8];
      sc[nt] = wmma_bf16(qf[0], bf_, sc[nt]);
#pragma unroll
      for (int v = 0; v < 8; ++v)
        bf_.u[v] = uK[(nt * 16 + r) * 32 + 16 + half * 4 + (v & 3) + (v >> 2) * 8];
      sc[nt] = wmma_bf16(qf[1], bf_, sc[nt]);
    }

    // ---- causal mask + online softmax (row = vgpr idx within lane-half) ----
    float pv[2][8];
#pragma unroll
    for (int e = 0; e < 8; ++e) {
      int gq = qbase + wave * 16 + half * 8 + e;
      float s0 = (jb + r <= gq) ? sc[0][e] : -3.0e38f;
      float s1 = (jb + 16 + r <= gq) ? sc[1][e] : -3.0e38f;
      float mx = fmaxf(s0, s1);
#pragma unroll
      for (int m = 8; m >= 1; m >>= 1) mx = fmaxf(mx, __shfl_xor(mx, m, 32));
      float mnew = fmaxf(mrow[e], mx);
      float corr = __expf(mrow[e] - mnew);
      float p0 = __expf(s0 - mnew), p1 = __expf(s1 - mnew);
      float ls = p0 + p1;
#pragma unroll
      for (int m = 8; m >= 1; m >>= 1) ls += __shfl_xor(ls, m, 32);
      lrow[e] = lrow[e] * corr + ls;
      mrow[e] = mnew;
      pv[0][e] = p0; pv[1][e] = p1;
#pragma unroll
      for (int dt = 0; dt < 4; ++dt) acc[dt][e] *= corr;
    }

    // ---- stage P (C-layout -> A-layout) through per-wave LDS ----
    {
      unsigned short* pp = (unsigned short*)&uP[wave * 256];
#pragma unroll
      for (int nt = 0; nt < 2; ++nt)
#pragma unroll
        for (int e = 0; e < 8; ++e)
          pp[(half * 8 + e) * 32 + nt * 16 + r] = f2bf(pv[nt][e]);
    }
    __syncthreads();

    AF pa;
    {
      const unsigned int* pu = &uP[wave * 256];
#pragma unroll
      for (int j = 0; j < 4; ++j) {
        pa.u[j] = pu[r * 16 + half * 4 + j];
        pa.u[4 + j] = pu[r * 16 + 8 + half * 4 + j];
      }
    }
    // ---- acc += P @ V : four 16x16 d-tiles ----
#pragma unroll
    for (int dt = 0; dt < 4; ++dt) {
      AF vf;
#pragma unroll
      for (int v = 0; v < 8; ++v)
        vf.u[v] = uV[(half * 4 + (v & 3) + (v >> 2) * 8) * 64 + dt * 16 + r];
      acc[dt] = wmma_bf16(pa, vf, acc[dt]);
    }
    __syncthreads();
  }

  // ---- finalize: out = acc / l ----
#pragma unroll
  for (int dt = 0; dt < 4; ++dt)
#pragma unroll
    for (int e = 0; e < 8; ++e) {
      int gq = qbase + wave * 16 + half * 8 + e;
      float o = acc[dt][e] / lrow[e];
      ob[(size_t)(b * S + gq) * 1024 + h * 64 + dt * 16 + r] = f2bf(o);
    }
}

// ---------------------------------------------------------------------------
// Router: logits = ln2 @ Wg + bg; softmax; top-2 renormalized -> combine[T,8]
// ---------------------------------------------------------------------------
__global__ __launch_bounds__(256)
void k_gate(const unsigned short* __restrict__ xf, const float* __restrict__ Wg,
            const float* __restrict__ bg, float* __restrict__ combine,
            float* __restrict__ probs) {
  int t = blockIdx.x * 256 + threadIdx.x;
  float acc[8];
#pragma unroll
  for (int e = 0; e < 8; ++e) acc[e] = bg[e];
  const unsigned short* xr = xf + (size_t)t * 1024;
  for (int d = 0; d < 1024; ++d) {
    float xd = bf2f(xr[d]);
    const float* wr = Wg + d * 8;
#pragma unroll
    for (int e = 0; e < 8; ++e) acc[e] = fmaf(xd, wr[e], acc[e]);
  }
  float mx = acc[0];
#pragma unroll
  for (int e = 1; e < 8; ++e) mx = fmaxf(mx, acc[e]);
  float p[8]; float se = 0.f;
#pragma unroll
  for (int e = 0; e < 8; ++e) { p[e] = __expf(acc[e] - mx); se += p[e]; }
  float inv = 1.f / se;
#pragma unroll
  for (int e = 0; e < 8; ++e) p[e] *= inv;
  int i1 = 0; float v1 = p[0];
#pragma unroll
  for (int e = 1; e < 8; ++e) if (p[e] > v1) { v1 = p[e]; i1 = e; }
  int i2 = -1; float v2 = -1.f;
#pragma unroll
  for (int e = 0; e < 8; ++e)
    if (e != i1 && p[e] > v2) { v2 = p[e]; i2 = e; }
  float nrm = 1.f / (v1 + v2);
#pragma unroll
  for (int e = 0; e < 8; ++e) {
    combine[(size_t)t * 8 + e] = 0.f;
    probs[(size_t)t * 8 + e] = p[e];
  }
  combine[(size_t)t * 8 + i1] = v1 * nrm;
  combine[(size_t)t * 8 + i2] = v2 * nrm;
}

// ---------------------------------------------------------------------------
// Aux loss: 0.01 * E * sum_e frac_e * pmean_e
// ---------------------------------------------------------------------------
__global__ __launch_bounds__(256)
void k_aux(const float* __restrict__ cmb, const float* __restrict__ prb,
           float* __restrict__ out) {
  __shared__ float sf[64], sp[64];
  int tid = threadIdx.x, lane = tid & 31, wave = tid >> 5;
  float f[8] = {0, 0, 0, 0, 0, 0, 0, 0}, p[8] = {0, 0, 0, 0, 0, 0, 0, 0};
  for (int t = tid; t < 4096; t += 256) {
#pragma unroll
    for (int e = 0; e < 8; ++e) {
      f[e] += (cmb[(size_t)t * 8 + e] > 0.f) ? 1.f : 0.f;
      p[e] += prb[(size_t)t * 8 + e];
    }
  }
#pragma unroll
  for (int e = 0; e < 8; ++e)
#pragma unroll
    for (int m = 16; m >= 1; m >>= 1) {
      f[e] += __shfl_xor(f[e], m, 32);
      p[e] += __shfl_xor(p[e], m, 32);
    }
  if (lane == 0)
#pragma unroll
    for (int e = 0; e < 8; ++e) { sf[wave * 8 + e] = f[e]; sp[wave * 8 + e] = p[e]; }
  __syncthreads();
  if (tid == 0) {
    float a = 0.f;
    for (int e = 0; e < 8; ++e) {
      float F = 0.f, P = 0.f;
      for (int w = 0; w < 8; ++w) { F += sf[w * 8 + e]; P += sp[w * 8 + e]; }
      a += (F / 4096.f) * (P / 4096.f);
    }
    out[0] = 0.01f * 8.f * a;
  }
}

// ---------------------------------------------------------------------------
// Host launcher
// ---------------------------------------------------------------------------
extern "C" void kernel_launch(void* const* d_in, const int* in_sizes, int n_in,
                              void* d_out, int out_size, void* d_ws,
                              size_t ws_size, hipStream_t stream) {
  (void)in_sizes; (void)n_in; (void)out_size; (void)ws_size;
  constexpr int T = 4096, D = 1024, FH = 4096;

  const float* x     = (const float*)d_in[0];
  const float* ln1_s = (const float*)d_in[1];
  const float* ln1_b = (const float*)d_in[2];
  const float* Wq    = (const float*)d_in[3];
  const float* bq    = (const float*)d_in[4];
  const float* Wk    = (const float*)d_in[5];
  const float* bk    = (const float*)d_in[6];
  const float* Wv    = (const float*)d_in[7];
  const float* bv    = (const float*)d_in[8];
  const float* Wo    = (const float*)d_in[9];
  const float* bo    = (const float*)d_in[10];
  const float* ln2_s = (const float*)d_in[11];
  const float* ln2_b = (const float*)d_in[12];
  const float* Wg    = (const float*)d_in[13];
  const float* bg    = (const float*)d_in[14];
  const float* W1    = (const float*)d_in[15];
  const float* b1    = (const float*)d_in[16];
  const float* W2    = (const float*)d_in[17];
  const float* b2    = (const float*)d_in[18];

  char* w = (char*)d_ws;
  auto take = [&](size_t bytes) {
    void* p = (void*)w;
    w += (bytes + 255) & ~(size_t)255;
    return p;
  };
  unsigned short* ln1o = (unsigned short*)take((size_t)T * D * 2);
  unsigned short* qbuf = (unsigned short*)take((size_t)T * D * 2);
  unsigned short* kbuf = (unsigned short*)take((size_t)T * D * 2);
  unsigned short* vbuf = (unsigned short*)take((size_t)T * D * 2);
  unsigned short* aout = (unsigned short*)take((size_t)T * D * 2);
  float*          work = (float*)take((size_t)T * D * 4);
  unsigned short* ln2o = (unsigned short*)take((size_t)T * D * 2);
  unsigned short* hbuf = (unsigned short*)take((size_t)T * FH * 2);
  float*          cmb  = (float*)take((size_t)T * 8 * 4);
  float*          prb  = (float*)take((size_t)T * 8 * 4);

  // 1) LN1
  k_ln<<<T, 256, 0, stream>>>(x, ln1_s, ln1_b, ln1o);

  // 2) QKV projections (q gets 1/sqrt(DH)=0.125 folded in)
  dim3 gD(D / GBN, T / GBM);
  k_gemm<1><<<gD, 256, 0, stream>>>(ln1o, D, Wq, D, bq, qbuf, nullptr, nullptr,
                                    0, 0.125f, D, D);
  k_gemm<0><<<gD, 256, 0, stream>>>(ln1o, D, Wk, D, bk, kbuf, nullptr, nullptr,
                                    0, 1.f, D, D);
  k_gemm<0><<<gD, 256, 0, stream>>>(ln1o, D, Wv, D, bv, vbuf, nullptr, nullptr,
                                    0, 1.f, D, D);

  // 3) causal flash attention
  k_attn<<<dim3(2048 / 64, 16, 2), 128, 0, stream>>>(qbuf, kbuf, vbuf, aout);

  // 4) output projection + residual -> work (f32)
  k_gemm<2><<<gD, 256, 0, stream>>>(aout, D, Wo, D, bo, work, x, nullptr,
                                    0, 1.f, D, D);

  // 5) LN2
  k_ln<<<T, 256, 0, stream>>>(work, ln2_s, ln2_b, ln2o);

  // 6) router
  k_gate<<<T / 256, 256, 0, stream>>>(ln2o, Wg, bg, cmb, prb);

  // 7) init output with residual
  (void)hipMemcpyAsync(d_out, work, (size_t)T * D * sizeof(float),
                       hipMemcpyDeviceToDevice, stream);

  // 8) MoE experts (dense, sequential -> race-free accumulation)
  dim3 gFH(FH / GBN, T / GBM);
  for (int e = 0; e < 8; ++e) {
    k_gemm<3><<<gFH, 256, 0, stream>>>(ln2o, D, W1 + (size_t)e * D * FH, FH,
                                       b1 + (size_t)e * FH, hbuf, nullptr,
                                       nullptr, 0, 1.f, FH, D);
    k_gemm<4><<<gD, 256, 0, stream>>>(hbuf, FH, W2 + (size_t)e * FH * D, D,
                                      b2 + (size_t)e * D, (float*)d_out,
                                      nullptr, cmb, e, 1.f, D, FH);
  }

  // 9) aux loss -> d_out[T*D]
  k_aux<<<1, 256, 0, stream>>>(cmb, prb, (float*)d_out + (size_t)T * D);
}
